// SequentialDPClustering_5781025980419
// MI455X (gfx1250) — matrix operand order repaired and checked by reading, not compiled, not run on hardware
//
#include <hip/hip_runtime.h>
#include <hip/hip_bf16.h>

#define TT 2048
#define DD 256
#define BB 8
#define KNN 10
#define BETA_C 0.2f
#define THRESH_C 0.7f
#define MAXSPAN_C 4

typedef __attribute__((ext_vector_type(16))) __bf16 v16bf;
typedef __attribute__((ext_vector_type(8)))  __bf16 v8bf;
typedef __attribute__((ext_vector_type(8)))  float  v8f;

union BfPack { v16bf v; v8bf h[2]; };

// A fragment (16x32, M x K), bf16, per CDNA5 ISA 7.12.2:
// lane: M = lane&15 (both halves), elements e: K = (e<8 ? e : e+8) + 8*(lane>>4)
__device__ __forceinline__ v16bf load_a_frag(const __bf16* xn, int row, int k0, int half) {
    const __bf16* p = xn + row * DD + k0 + half * 8;
    BfPack r;
    r.h[0] = *reinterpret_cast<const v8bf*>(p);        // K = half*8 + 0..7
    r.h[1] = *reinterpret_cast<const v8bf*>(p + 16);   // K = half*8 + 16..23
    return r.v;
}

// B fragment (32x16, K x N), bf16: lane: N = lane&15, elements e: K = (e<8 ? e : e+8) + 8*(lane>>4)
__device__ __forceinline__ v16bf load_b_frag(const __bf16* xn, int row, int k0, int half) {
    const __bf16* p = xn + row * DD + k0 + half * 8;
    BfPack r;
    r.h[0] = *reinterpret_cast<const v8bf*>(p);
    r.h[1] = *reinterpret_cast<const v8bf*>(p + 16);
    return r.v;
}

// Preload all 8 K-step B fragments of one 16-column tile into distinct regs.
__device__ __forceinline__ void load_b_tile(const __bf16* xb, int j0, int mr, int half,
                                            v16bf bf[8]) {
    #pragma unroll
    for (int kk = 0; kk < 8; ++kk) bf[kk] = load_b_frag(xb, j0 + mr, kk * 32, half);
}

// ---------------- kernel 1: L2-normalize rows, convert to bf16 [B,T,D] -------
__global__ void dp_normalize(const float* __restrict__ x, __bf16* __restrict__ xn) {
    int bt = blockIdx.x;               // b*T + t
    int b = bt / TT, t = bt % TT;
    int d = threadIdx.x;               // blockDim = 256 = D
    __shared__ float red[DD];
    float v = x[((size_t)b * DD + d) * TT + t];
    red[d] = v * v;
    __syncthreads();
    for (int off = 128; off > 0; off >>= 1) {
        if (d < off) red[d] += red[d + off];
        __syncthreads();
    }
    float nrm = fmaxf(sqrtf(red[0]), 1e-12f);
    xn[((size_t)b * TT + t) * DD + d] = (__bf16)(v / nrm);
}

// -------- kernel 2: WMMA sim sweep -> rho (top-K NN density), rowmax, band ---
__global__ void __launch_bounds__(32) __attribute__((amdgpu_waves_per_eu(1)))
dp_stats(const __bf16* __restrict__ xn,
         float* __restrict__ rho_g, float* __restrict__ rmax_g,
         float* __restrict__ band_g) {
    int blk = blockIdx.x;
    int b  = blk / (TT / 16);
    int i0 = (blk % (TT / 16)) * 16;
    const __bf16* xb = xn + (size_t)b * TT * DD;
    int lane = threadIdx.x;            // blockDim = 32 (one wave)
    int half = lane >> 4, mr = lane & 15;

    v16bf a[8];
    #pragma unroll
    for (int kk = 0; kk < 8; ++kk) a[kk] = load_a_frag(xb, i0 + mr, kk * 32, half);

    // per-lane partial stats: lane owns (row = mr, columns half*8..half*8+7)
    float top[KNN + 1];
    #pragma unroll
    for (int q = 0; q <= KNN; ++q) top[q] = -1e30f;
    float rmax = -1e30f;
    int i = i0 + mr;                   // row this lane scans in the epilogue

    __shared__ float tile[16][17];
    __shared__ float mtop[32][KNN + 2];
    __shared__ float mmax[32];

    for (int j0 = 0; j0 < TT; j0 += 16) {
        v16bf bfr[8];
        load_b_tile(xb, j0, mr, half, bfr);    // 16 b128 loads
        __builtin_amdgcn_sched_barrier(0);     // keep loads grouped before WMMAs
        v8f acc = {};
        #pragma unroll
        for (int kk = 0; kk < 8; ++kk)
            acc = __builtin_amdgcn_wmma_f32_16x16x32_bf16(false, a[kk], false, bfr[kk],
                                                          (short)0, acc, false, false);
        #pragma unroll
        for (int r = 0; r < 8; ++r) tile[r + 8 * half][mr] = acc[r]; // C: M=r+8*half, N=mr
        __syncthreads();
        // all 32 lanes: 8 columns each of row mr
        #pragma unroll
        for (int nn = 0; nn < 8; ++nn) {
            int n = half * 8 + nn;
            int j = j0 + n;
            float sim = (tile[mr][n] + 1.0f) * 0.5f;
            rmax = fmaxf(rmax, sim);
            int dd = j - i;
            if (dd >= -MAXSPAN_C && dd <= MAXSPAN_C)
                band_g[((size_t)b * TT + i) * 9 + (dd + 4)] = sim;
            if (sim > top[KNN]) {      // threshold-guarded ripple insert
                float v = sim;
                #pragma unroll
                for (int q = 0; q <= KNN; ++q) {
                    float o = top[q];
                    bool g = v > o;
                    top[q] = g ? v : o;
                    v      = g ? o : v;
                }
            }
        }
        __syncthreads();
    }

    // merge the two half-row partials (lane l and l+16 share row l&15)
    #pragma unroll
    for (int q = 0; q <= KNN; ++q) mtop[lane][q] = top[q];
    mmax[lane] = rmax;
    __syncthreads();
    if (lane < 16) {
        #pragma unroll
        for (int q = 0; q <= KNN; ++q) {
            float v = mtop[lane + 16][q];
            if (v > top[KNN]) {
                #pragma unroll
                for (int p = 0; p <= KNN; ++p) {
                    float o = top[p];
                    bool g = v > o;
                    top[p] = g ? v : o;
                    v      = g ? o : v;
                }
            }
        }
        float m = 0.f;
        #pragma unroll
        for (int q = 1; q <= KNN; ++q) m += top[q];  // drop largest (self)
        rho_g[b * TT + i]  = __expf(-m / (float)KNN);
        rmax_g[b * TT + i] = fmaxf(rmax, mmax[lane + 16]);
    }
}

// -------- kernel 3: WMMA sim sweep -> delta (min sim to higher-rho), s -------
__global__ void __launch_bounds__(32) __attribute__((amdgpu_waves_per_eu(1)))
dp_delta(const __bf16* __restrict__ xn,
         const float* __restrict__ rho_g,
         const float* __restrict__ rmax_g,
         float* __restrict__ s_g) {
    int blk = blockIdx.x;
    int b  = blk / (TT / 16);
    int i0 = (blk % (TT / 16)) * 16;
    const __bf16* xb = xn + (size_t)b * TT * DD;
    int lane = threadIdx.x;
    int half = lane >> 4, mr = lane & 15;

    v16bf a[8];
    #pragma unroll
    for (int kk = 0; kk < 8; ++kk) a[kk] = load_a_frag(xb, i0 + mr, kk * 32, half);

    int i = i0 + mr;
    float ri = rho_g[b * TT + i];      // loop-invariant row density
    float dmin = 3.0e38f;
    __shared__ float tile[16][17];
    __shared__ float rhoj[16];
    __shared__ float mdmin[32];

    for (int j0 = 0; j0 < TT; j0 += 16) {
        v16bf bfr[8];
        load_b_tile(xb, j0, mr, half, bfr);
        __builtin_amdgcn_sched_barrier(0);
        v8f acc = {};
        #pragma unroll
        for (int kk = 0; kk < 8; ++kk)
            acc = __builtin_amdgcn_wmma_f32_16x16x32_bf16(false, a[kk], false, bfr[kk],
                                                          (short)0, acc, false, false);
        #pragma unroll
        for (int r = 0; r < 8; ++r) tile[r + 8 * half][mr] = acc[r];
        if (lane < 16) rhoj[lane] = rho_g[b * TT + j0 + lane];
        __syncthreads();
        #pragma unroll
        for (int nn = 0; nn < 8; ++nn) {
            int n = half * 8 + nn;
            float sim = (tile[mr][n] + 1.0f) * 0.5f;
            if (rhoj[n] > ri) dmin = fminf(dmin, sim);
        }
        __syncthreads();
    }

    mdmin[lane] = dmin;
    __syncthreads();
    if (lane < 16) {
        float dm = fminf(dmin, mdmin[lane + 16]);
        float delta = (dm > 1e37f) ? rmax_g[b * TT + i] : dm;
        s_g[b * TT + i] = ri * delta;
    }
}

// -------- kernel 4: greedy sequential density-peak clustering (1 block/batch)
__global__ void dp_cluster(const float* __restrict__ s_g,
                           const float* __restrict__ band_g,
                           int* __restrict__ starts_g, int* __restrict__ cnts_g,
                           int* __restrict__ numc_g) {
    int b = blockIdx.x;
    int tid = threadIdx.x;             // blockDim = 256
    __shared__ float s_sh[TT];
    __shared__ int asn[TT];
    __shared__ int lab[TT];
    __shared__ float bv[256];
    __shared__ int bi[256];
    __shared__ int cnt;

    for (int t = tid; t < TT; t += 256) {
        s_sh[t] = s_g[b * TT + t];
        asn[t] = 0;
        lab[t] = 0;
    }
    if (tid == 0) cnt = 0;
    __syncthreads();

    for (;;) {
        __syncthreads();
        if (cnt >= TT) break;
        // first-index argmax over unassigned s
        float best = -3e38f; int bidx = 0;
        for (int t = tid; t < TT; t += 256) {
            float v = asn[t] ? -1e9f : s_sh[t];
            if (v > best) { best = v; bidx = t; }   // strict > keeps earliest t
        }
        bv[tid] = best; bi[tid] = bidx;
        __syncthreads();
        for (int off = 128; off > 0; off >>= 1) {
            if (tid < off) {
                float v2 = bv[tid + off]; int i2 = bi[tid + off];
                if (v2 > bv[tid] || (v2 == bv[tid] && i2 < bi[tid])) { bv[tid] = v2; bi[tid] = i2; }
            }
            __syncthreads();
        }
        if (tid == 0) {
            int seed = bi[0];
            asn[seed] = 1; lab[seed] = seed; cnt += 1;
            int size = 1;
            for (int dir = 0; dir < 2; ++dir) {      // forward then backward
                int st = (dir == 0) ? 1 : -1;
                bool ok = true;
                for (int d = 1; d <= MAXSPAN_C; ++d) {
                    int t  = seed + st * d;
                    bool inb = (t >= 0) && (t < TT);
                    int tc = t < 0 ? 0 : (t > TT - 1 ? TT - 1 : t);
                    bool inc = ok && inb && (asn[tc] == 0) && (size < MAXSPAN_C) &&
                               (band_g[((size_t)b * TT + seed) * 9 + (st * d + 4)]
                                - BETA_C * s_sh[tc] > THRESH_C);
                    if (inc) { asn[tc] = 1; lab[tc] = seed; ++size; ++cnt; }
                    ok = inc;
                }
            }
        }
    }
    __syncthreads();
    if (tid == 0) {   // contiguous runs -> ordinal cluster table
        int nc = 0, runStart = 0;
        for (int t = 1; t < TT; ++t) {
            if (lab[t] != lab[t - 1]) {
                starts_g[b * TT + nc] = runStart;
                cnts_g[b * TT + nc]   = t - runStart;
                ++nc; runStart = t;
            }
        }
        starts_g[b * TT + nc] = runStart;
        cnts_g[b * TT + nc]   = TT - runStart;
        ++nc;
        numc_g[b] = nc;
    }
}

// -------- kernel 5: mean-pool frames per cluster -> embs [B,D,T], lengths ----
__global__ void dp_aggregate(const float* __restrict__ x,
                             const int* __restrict__ starts_g,
                             const int* __restrict__ cnts_g,
                             const int* __restrict__ numc_g,
                             float* __restrict__ embs, float* __restrict__ lens) {
    int c = blockIdx.x, b = blockIdx.y, d = threadIdx.x;   // blockDim = 256 = D
    int nc = numc_g[b];
    float val = 0.f, cn = 0.f;
    if (c < nc) {
        int st = starts_g[b * TT + c];
        int k  = cnts_g[b * TT + c];
        cn = (float)k;
        const float* xp = x + ((size_t)b * DD + d) * TT + st;
        for (int i = 0; i < k; ++i) val += xp[i];
        val /= cn;
    }
    embs[((size_t)b * DD + d) * TT + c] = val;
    if (d == 0) lens[(size_t)b * TT + c] = cn;
}

extern "C" void kernel_launch(void* const* d_in, const int* in_sizes, int n_in,
                              void* d_out, int out_size, void* d_ws, size_t ws_size,
                              hipStream_t stream) {
    (void)in_sizes; (void)n_in; (void)out_size; (void)ws_size;
    const float* x = (const float*)d_in[0];

    float* out  = (float*)d_out;
    float* embs = out;                               // B*D*T floats
    float* lens = out + (size_t)BB * DD * TT;        // B*T floats

    // bf16 normalized copy staged in d_out (8 MB < 16.8 MB); overwritten only
    // by the final aggregation kernel, which reads x + cluster tables only.
    __bf16* xn = (__bf16*)d_out;

    // small scratch in d_ws (~0.9 MB total)
    char* w = (char*)d_ws;
    float* rho  = (float*)w;  w += (size_t)BB * TT * 4;
    float* rmax = (float*)w;  w += (size_t)BB * TT * 4;
    float* s    = (float*)w;  w += (size_t)BB * TT * 4;
    float* band = (float*)w;  w += (size_t)BB * TT * 9 * 4;
    int* starts = (int*)w;    w += (size_t)BB * TT * 4;
    int* cnts   = (int*)w;    w += (size_t)BB * TT * 4;
    int* numc   = (int*)w;    w += (size_t)BB * 4;

    dp_normalize<<<BB * TT, DD, 0, stream>>>(x, xn);
    dp_stats<<<BB * (TT / 16), 32, 0, stream>>>(xn, rho, rmax, band);
    dp_delta<<<BB * (TT / 16), 32, 0, stream>>>(xn, rho, rmax, s);
    dp_cluster<<<BB, 256, 0, stream>>>(s, band, starts, cnts, numc);
    dp_aggregate<<<dim3(TT, BB), DD, 0, stream>>>(x, starts, cnts, numc, embs, lens);
}
